// SchNetInteraction_7602092114194
// MI455X (gfx1250) — matrix operand order repaired and checked by reading, not compiled, not run on hardware
//
#include <hip/hip_runtime.h>
#include <math.h>

typedef __attribute__((ext_vector_type(16))) _Float16 v16h;
typedef __attribute__((ext_vector_type(4)))  _Float16 v4h;
typedef __attribute__((ext_vector_type(8)))  float    v8f;
typedef int v4i __attribute__((vector_size(16)));

#define ADIM 1024
#define NNBR 64
#define SDIM 50
#define FDIM 128

#define HAVE_ASYNC_LDS (__has_builtin(__builtin_amdgcn_global_load_async_to_lds_b128) && \
                        __has_builtin(__builtin_amdgcn_s_wait_asynccnt))

// shifted softplus, numerically stable
__device__ __forceinline__ float sspf(float x) {
    return fmaxf(x, 0.0f) + log1pf(expf(-fabsf(x))) - 0.69314718055994531f;
}

// ---------- fragment-order ("swizzled") offsets, in halves ----------
// A 16x32 f16 (ISA 7.12.2): lane = (m&15) + 16*((k>>3)&1), half = (k&7) + 8*((k>>4)&1)
__device__ __forceinline__ int a_swz(int m, int k, int nkt) {
    int frag = (m >> 4) * nkt + (k >> 5);
    int lane = (m & 15) + (((k >> 3) & 1) << 4);
    int i    = (k & 7) + (((k >> 4) & 1) << 3);
    return (frag * 32 + lane) * 16 + i;
}
// B 32x16 f16: lane = (n&15) + 16*((k>>3)&1), half = (k&7) + 8*((k>>4)&1). 8 N-tiles across 128 cols.
__device__ __forceinline__ int b_swz(int k, int n) {
    int frag = (k >> 5) * 8 + (n >> 4);
    int lane = (n & 15) + (((k >> 3) & 1) << 4);
    int i    = (k & 7) + (((k >> 4) & 1) << 3);
    return (frag * 32 + lane) * 16 + i;
}

// ---------- weight image -> LDS (async DMA path; vector-copy fallback) ----------
__device__ __forceinline__ void copy_w_lds(const _Float16* __restrict__ g,
                                           _Float16* s, int halves, int tid) {
#if HAVE_ASYNC_LDS
    v4i __attribute__((address_space(1)))* gp =
        (v4i __attribute__((address_space(1)))*)(uintptr_t)g;
    v4i __attribute__((address_space(3)))* lp =
        (v4i __attribute__((address_space(3)))*)(uint32_t)(uintptr_t)s;
    const int n16 = halves / 8;                  // 16-byte packets
    for (int i = tid; i < n16; i += 256)
        __builtin_amdgcn_global_load_async_to_lds_b128(gp + i, lp + i, 0, 0);
#else
    const uint4* g4 = (const uint4*)g;
    uint4* s4 = (uint4*)s;
    for (int i = tid; i < halves / 8; i += 256) s4[i] = g4[i];
#endif
}
__device__ __forceinline__ void async_fence() {
#if HAVE_ASYNC_LDS
    __builtin_amdgcn_s_wait_asynccnt(0);
#endif
}

// ---------- staging: global f32 -> swizzled f16 LDS (activations) ----------
template<int KPAD>
__device__ __forceinline__ void stage_a_pad(const float* __restrict__ g, int ksrc,
                                            _Float16* sAf, int tid) {
    constexpr int NKT = KPAD / 32, KQ = KPAD / 4;
    for (int idx = tid; idx < 64 * KQ; idx += 256) {
        int m = idx / KQ, k = (idx - m * KQ) * 4;
        v4h p;
#pragma unroll
        for (int t = 0; t < 4; ++t)
            p[t] = (k + t < ksrc) ? (_Float16)g[m * ksrc + k + t] : (_Float16)0.0f;
        *(v4h*)(sAf + a_swz(m, k, NKT)) = p;
    }
}
template<int KPAD>
__device__ __forceinline__ void stage_a_full(const float* __restrict__ g,
                                             _Float16* sAf, int tid) {
    constexpr int NKT = KPAD / 32, KQ = KPAD / 4;
    for (int idx = tid; idx < 64 * KQ; idx += 256) {
        int m = idx / KQ, k = (idx - m * KQ) * 4;
        const float4 f = *(const float4*)(g + m * KPAD + k);
        v4h p;
        p[0] = (_Float16)f.x; p[1] = (_Float16)f.y;
        p[2] = (_Float16)f.z; p[3] = (_Float16)f.w;
        *(v4h*)(sAf + a_swz(m, k, NKT)) = p;
    }
}

// ---------- 64x128 WMMA tile engine: 8 waves * 4 D-tiles, fully unrolled ----------
template<int NKT>
__device__ __forceinline__ void gemm_64x128(const _Float16* sAf, const _Float16* sBf,
                                            v8f acc[4], int wave, int lane) {
    const int mtile = wave & 3;
    const int nt0   = (wave >> 2) << 2;
#pragma unroll
    for (int kt = 0; kt < NKT; ++kt) {
        v16h a = *(const v16h*)(sAf + ((mtile * NKT + kt) * 32 + lane) * 16);
#pragma unroll
        for (int j = 0; j < 4; ++j) {
            v16h b = *(const v16h*)(sBf + (((kt << 3) + nt0 + j) * 32 + lane) * 16);
            acc[j] = __builtin_amdgcn_wmma_f32_16x16x32_f16(
                false, a, false, b, (short)0, acc[j], false, false);
        }
    }
}

#define EPILOGUE_FOR(wave, lane, ROW, COL, BODY)                         \
    {                                                                    \
        const int m0_ = ((wave) & 3) << 4, nb_ = ((wave) >> 2) << 6;     \
        const int hh_ = (lane) >> 4, nl_ = (lane) & 15;                  \
        _Pragma("unroll") for (int j = 0; j < 4; ++j)                    \
        _Pragma("unroll") for (int r = 0; r < 8; ++r) {                  \
            const int ROW = m0_ + r + (hh_ << 3);                        \
            const int COL = nb_ + (j << 4) + nl_;                        \
            BODY                                                         \
        }                                                                \
    }

// ===== Kernel 0: one-time f32 -> fragment-swizzled f16 weight images =====
__global__ __launch_bounds__(256) void k_prep(const float* __restrict__ Wf1,
                                              const float* __restrict__ Wf2,
                                              const float* __restrict__ Win2f,
                                              const float* __restrict__ Wf2o,
                                              const float* __restrict__ Wd,
                                              _Float16* __restrict__ o) {
    const int tid = threadIdx.x, wsel = blockIdx.x;
    const float* src = (wsel == 0) ? Wf1 : (wsel == 1) ? Wf2 :
                       (wsel == 2) ? Win2f : (wsel == 3) ? Wf2o : Wd;
    _Float16* dst = o + (size_t)wsel * 16384;
    const int krows = (wsel == 0) ? SDIM : FDIM;
    const int kqn   = (wsel == 0) ? 16 : 32;       // ktiles*8 quads
    const int n = tid & 127;
    for (int kq = tid >> 7; kq < kqn; kq += 2) {
        int k = kq * 4;
        v4h p;
#pragma unroll
        for (int t = 0; t < 4; ++t)
            p[t] = (k + t < krows) ? (_Float16)src[(k + t) * FDIM + n] : (_Float16)0.0f;
        *(v4h*)(dst + b_swz(k, n)) = p;
    }
}

// ================= Kernel 1: y = x @ W_in2f =================
__global__ __launch_bounds__(256) void k_in2f(const float* __restrict__ x,
                                              const _Float16* __restrict__ Wsw,
                                              float* __restrict__ y) {
    __shared__ alignas(32) _Float16 sAf[64 * 128];
    __shared__ alignas(32) _Float16 sBf[128 * 128];
    const int tid = threadIdx.x, lane = tid & 31, wave = tid >> 5;
    const int row0 = blockIdx.x * 64;

    copy_w_lds(Wsw, sBf, 16384, tid);
    stage_a_full<128>(x + (size_t)row0 * FDIM, sAf, tid);
    async_fence();
    __syncthreads();

    v8f acc[4] = {};
    gemm_64x128<4>(sAf, sBf, acc, wave, lane);

    EPILOGUE_FOR(wave, lane, row, col, {
        y[(size_t)(row0 + row) * FDIM + col] = acc[j][r];
    })
}

// ==== Kernel 2: fused filter-network + cutoff + gather + neighbor reduction ====
__global__ __launch_bounds__(256) void k_filter_conv(
    const float* __restrict__ r_ij, const int* __restrict__ neighbors,
    const float* __restrict__ mask, const float* __restrict__ f_ij,
    const _Float16* __restrict__ Wf1sw, const float* __restrict__ bf1,
    const _Float16* __restrict__ Wf2sw, const float* __restrict__ bf2,
    const float* __restrict__ y, float* __restrict__ ybar) {
    __shared__ alignas(32) _Float16 sAf[64 * 64];    // f_ij, K padded 50->64
    __shared__ alignas(32) _Float16 sHf[64 * 128];   // hidden acts (A-swizzled)
    __shared__ alignas(32) _Float16 sBf[128 * 128];  // Wf1 then Wf2; finally f32 W
    __shared__ float sC[64];
    __shared__ float sB1[128], sB2[128];
    __shared__ int   sNb[64];
    __shared__ float sRed[256];

    const int tid = threadIdx.x, lane = tid & 31, wave = tid >> 5;
    const int ba = blockIdx.x;                       // b*A + a
    const float* fij = f_ij + (size_t)ba * NNBR * SDIM;

    __builtin_prefetch(Wf2sw + (size_t)tid * 64, 0, 1);   // pull Wf2 image into L2

    copy_w_lds(Wf1sw, sBf, 8192, tid);               // 2 ktiles only (K=64)
    stage_a_pad<64>(fij, SDIM, sAf, tid);
    if (tid < NNBR) {
        float rr = r_ij[(size_t)ba * NNBR + tid];
        float cc = 0.5f * (cosf(rr * (3.14159265358979f / 5.0f)) + 1.0f);
        cc = (rr < 5.0f) ? cc : 0.0f;
        sC[tid]  = cc * mask[(size_t)ba * NNBR + tid];
        sNb[tid] = neighbors[(size_t)ba * NNBR + tid];
    }
    if (tid < FDIM) { sB1[tid] = bf1[tid]; sB2[tid] = bf2[tid]; }
    async_fence();
    __syncthreads();

    // GEMM1: H = ssp(f_ij @ Wf1 + bf1), K=64
    v8f acc[4] = {};
    gemm_64x128<2>(sAf, sBf, acc, wave, lane);
    EPILOGUE_FOR(wave, lane, row, col, {
        sHf[a_swz(row, col, 4)] = (_Float16)sspf(acc[j][r] + sB1[col]);
    })
    __syncthreads();                                 // GEMM1 done reading sBf

    copy_w_lds(Wf2sw, sBf, 16384, tid);
    async_fence();
    __syncthreads();

    // GEMM2: W = (H @ Wf2 + bf2) * C, K=128
    v8f acc2[4] = {};
    gemm_64x128<4>(sHf, sBf, acc2, wave, lane);
    __syncthreads();                                 // all reads of sBf done
    float* sW = (float*)sBf;                         // 64x128 f32 fits exactly
    EPILOGUE_FOR(wave, lane, row, col, {
        sW[row * FDIM + col] = (acc2[j][r] + sB2[col]) * sC[row];
    })
    __syncthreads();

    // gather y rows by neighbor index + masked sum over N
    const int b = ba >> 10;                          // / ADIM
    const int f = tid & 127, halfn = tid >> 7;
    float s = 0.0f;
    for (int n = halfn * 32; n < halfn * 32 + 32; ++n) {
        int an = sNb[n];
        s += y[((size_t)(b * ADIM + an)) * FDIM + f] * sW[n * FDIM + f];
    }
    sRed[tid] = s;
    __syncthreads();
    if (tid < FDIM)
        ybar[(size_t)ba * FDIM + tid] = sRed[tid] + sRed[tid + FDIM];
}

// ===== Kernel 3: v = ssp(ybar @ W_f2out + b) @ W_dense + b_dense =====
__global__ __launch_bounds__(256) void k_out(
    const float* __restrict__ ybar, const _Float16* __restrict__ W1sw,
    const float* __restrict__ b1,   const _Float16* __restrict__ W2sw,
    const float* __restrict__ b2,   float* __restrict__ out) {
    __shared__ alignas(32) _Float16 sAf[64 * 128];   // input tile, reused for ssp()
    __shared__ alignas(32) _Float16 sBf[128 * 128];
    __shared__ float sB1[128], sB2[128];
    const int tid = threadIdx.x, lane = tid & 31, wave = tid >> 5;
    const int row0 = blockIdx.x * 64;

    copy_w_lds(W1sw, sBf, 16384, tid);
    stage_a_full<128>(ybar + (size_t)row0 * FDIM, sAf, tid);
    if (tid < FDIM) { sB1[tid] = b1[tid]; sB2[tid] = b2[tid]; }
    async_fence();
    __syncthreads();

    v8f acc[4] = {};
    gemm_64x128<4>(sAf, sBf, acc, wave, lane);
    __syncthreads();                                 // all done reading sAf & sBf
    EPILOGUE_FOR(wave, lane, row, col, {
        sAf[a_swz(row, col, 4)] = (_Float16)sspf(acc[j][r] + sB1[col]);
    })
    copy_w_lds(W2sw, sBf, 16384, tid);
    async_fence();
    __syncthreads();

    v8f acc2[4] = {};
    gemm_64x128<4>(sAf, sBf, acc2, wave, lane);
    EPILOGUE_FOR(wave, lane, row, col, {
        out[(size_t)(row0 + row) * FDIM + col] = acc2[j][r] + sB2[col];
    })
}

extern "C" void kernel_launch(void* const* d_in, const int* in_sizes, int n_in,
                              void* d_out, int out_size, void* d_ws, size_t ws_size,
                              hipStream_t stream) {
    (void)in_sizes; (void)n_in; (void)out_size; (void)ws_size;
    const float* x      = (const float*)d_in[0];
    const float* r_ij   = (const float*)d_in[1];
    const int*   nbrs   = (const int*)  d_in[2];
    const float* mask   = (const float*)d_in[3];
    const float* f_ij   = (const float*)d_in[4];
    const float* Wf1    = (const float*)d_in[5];
    const float* bf1    = (const float*)d_in[6];
    const float* Wf2    = (const float*)d_in[7];
    const float* bf2    = (const float*)d_in[8];
    const float* W_in2f = (const float*)d_in[9];
    const float* W_f2o  = (const float*)d_in[10];
    const float* b_f2o  = (const float*)d_in[11];
    const float* W_d    = (const float*)d_in[12];
    const float* b_d    = (const float*)d_in[13];

    float*    y_ws = (float*)d_ws;                   // [4096,128] f32 (2 MB)
    float*    ybar = y_ws + 4096 * FDIM;             // [4096,128] f32 (2 MB)
    _Float16* wsw  = (_Float16*)(ybar + 4096 * FDIM);// 5 x 16384 halves (160 KB)
    _Float16* wWf1   = wsw;
    _Float16* wWf2   = wsw + 1 * 16384;
    _Float16* wWin2f = wsw + 2 * 16384;
    _Float16* wWf2o  = wsw + 3 * 16384;
    _Float16* wWd    = wsw + 4 * 16384;
    float* out = (float*)d_out;

    k_prep<<<5, 256, 0, stream>>>(Wf1, Wf2, W_in2f, W_f2o, W_d, wsw);
    k_in2f<<<64, 256, 0, stream>>>(x, wWin2f, y_ws);
    k_filter_conv<<<4096, 256, 0, stream>>>(r_ij, nbrs, mask, f_ij,
                                            wWf1, bf1, wWf2, bf2, y_ws, ybar);
    k_out<<<64, 256, 0, stream>>>(ybar, wWf2o, b_f2o, wWd, b_d, out);
}
